// TemporalGraphConv_40407052321097
// MI455X (gfx1250) — matrix operand order repaired and checked by reading, not compiled, not run on hardware
//
#include <hip/hip_runtime.h>

typedef __attribute__((ext_vector_type(16))) __bf16 v16bf;
typedef __attribute__((ext_vector_type(8)))  __bf16 v8bf;
typedef __attribute__((ext_vector_type(8)))  float  v8f;

static constexpr int Tt   = 10;
static constexpr int Nn   = 20000;
static constexpr int Ee   = 320000;
static constexpr int INC  = 128;    // input channels
static constexpr int HIDc = 64;
static constexpr int OUTc = 128;
static constexpr int NH   = 4;      // heads
static constexpr int KDIM = NH * HIDc; // 256 columns of W_gat

// ---------------- conversion kernels ----------------

__global__ void k_convW(const float* __restrict__ W, __bf16* __restrict__ WT) {
    int i = blockIdx.x * blockDim.x + threadIdx.x;      // over 128*256
    if (i < INC * KDIM) {
        int k = i / KDIM, c = i % KDIM;
        WT[c * INC + k] = (__bf16)W[i];                 // transpose: [256][128]
    }
}

__global__ void k_convX(const float* __restrict__ x, __bf16* __restrict__ xb, int n) {
    int i = blockIdx.x * blockDim.x + threadIdx.x;
    if (i < n) xb[i] = (__bf16)x[i];
}

__global__ void k_init(float* __restrict__ acc, float* __restrict__ z) {
    int i = blockIdx.x * blockDim.x + threadIdx.x;
    if (i < Nn * KDIM) acc[i] = 0.0f;
    if (i < Nn * NH)   z[i]   = 0.0f;
}

// ---- WMMA GEMM: C[N,256] = Xbf[N,128] @ W[128,256] (bf16 in, f32 acc) ----
// One wave computes a 16x64 strip (4 accumulators, 16 v_wmma).  K-chunks are
// software-pipelined: chunk kc+1's A + 4xB fragments are prefetched into a
// second register set before chunk kc's WMMAs issue, so the matrix pipe never
// waits on a load that was issued immediately before it.

__device__ __forceinline__ v16bf load_afrag(const __bf16* __restrict__ arow,
                                            int kb, int half) {
    // A 16x32 bf16 fragment: half h holds contiguous K base+16h..base+16h+15
    v8bf lo = *(const v8bf*)(arow + kb + half * 16);
    v8bf hi = *(const v8bf*)(arow + kb + half * 16 + 8);
    v16bf a;
#pragma unroll
    for (int i = 0; i < 8; ++i) { a[i] = lo[i]; a[i + 8] = hi[i]; }
    return a;
}

__global__ void __launch_bounds__(32) k_gemm(const __bf16* __restrict__ X,
                                             const __bf16* __restrict__ WT,
                                             float* __restrict__ C) {
    int lane = threadIdx.x;            // wave32, one wave per block
    int r    = lane & 15;
    int half = lane >> 4;
    int m0 = blockIdx.x * 16;          // row tile
    int n0 = blockIdx.y * 64;          // 64-wide column strip

    const __bf16* arow  = X  + (size_t)(m0 + r) * INC;  // A: lane = row
    const __bf16* brow0 = WT + (size_t)(n0 + r) * INC;  // B: lane = col per sub-tile

    v8f c0 = {}, c1 = {}, c2 = {}, c3 = {};

    v16bf aF[2];
    v16bf bF[2][4];

    // prologue: chunk 0 fragments
    aF[0] = load_afrag(arow, 0, half);
#pragma unroll
    for (int j = 0; j < 4; ++j)
        bF[0][j] = *(const v16bf*)(brow0 + j * 16 * INC + half * 16);

#pragma unroll
    for (int kc = 0; kc < 4; ++kc) {
        const int cur = kc & 1, nxt = cur ^ 1;
        if (kc < 3) {                  // prefetch next chunk before consuming current
            int kb = (kc + 1) * 32;
            aF[nxt] = load_afrag(arow, kb, half);
#pragma unroll
            for (int j = 0; j < 4; ++j)
                bF[nxt][j] = *(const v16bf*)(brow0 + j * 16 * INC + kb + half * 16);
        }
        c0 = __builtin_amdgcn_wmma_f32_16x16x32_bf16(false, aF[cur], false, bF[cur][0], (short)0, c0, false, false);
        c1 = __builtin_amdgcn_wmma_f32_16x16x32_bf16(false, aF[cur], false, bF[cur][1], (short)0, c1, false, false);
        c2 = __builtin_amdgcn_wmma_f32_16x16x32_bf16(false, aF[cur], false, bF[cur][2], (short)0, c2, false, false);
        c3 = __builtin_amdgcn_wmma_f32_16x16x32_bf16(false, aF[cur], false, bF[cur][3], (short)0, c3, false, false);
    }

    // C/D layout: lane = col r, VGPR v = row v + 8*half
#pragma unroll
    for (int v = 0; v < 8; ++v) {
        float* crow = C + (size_t)(m0 + v + 8 * half) * KDIM + n0 + r;
        crow[0]  = c0[v];
        crow[16] = c1[v];
        crow[32] = c2[v];
        crow[48] = c3[v];
    }
}

// ---------------- attention-logit projections ----------------

__global__ void k_al(const float* __restrict__ xW, const float* __restrict__ a_src,
                     const float* __restrict__ a_dst,
                     float* __restrict__ al_s, float* __restrict__ al_d) {
    int i = blockIdx.x * blockDim.x + threadIdx.x;      // over N*4
    if (i >= Nn * NH) return;
    int h = i & 3;
    const float* row = xW + (size_t)(i >> 2) * KDIM + h * HIDc;
    const float* as  = a_src + h * HIDc;
    const float* ad  = a_dst + h * HIDc;
    float ss = 0.f, sd = 0.f;
#pragma unroll 8
    for (int c = 0; c < HIDc; ++c) { float v = row[c]; ss += v * as[c]; sd += v * ad[c]; }
    al_s[i] = ss; al_d[i] = sd;
}

// ---------------- edge softmax: numerator + denominator ----------------

__global__ void k_esum(const int* __restrict__ src, const int* __restrict__ dst,
                       const float* __restrict__ al_s, const float* __restrict__ al_d,
                       float* __restrict__ exbuf, float* __restrict__ z) {
    int i = blockIdx.x * blockDim.x + threadIdx.x;      // over (E+N)*4
    if (i >= (Ee + Nn) * NH) return;
    int e = i >> 2, h = i & 3;
    int s, d;
    if (e < Ee) { s = src[e]; d = dst[e]; } else { s = d = e - Ee; }  // self loops
    float f  = al_s[s * NH + h] + al_d[d * NH + h];
    float lr = f > 0.f ? f : 0.2f * f;                  // leaky_relu(0.2)
    float ex = __expf(lr);                              // logits are O(1): no max pass needed
    exbuf[i] = ex;
    atomicAdd(&z[d * NH + h], ex);
}

__global__ void k_alpha(const int* __restrict__ dst, const float* __restrict__ z,
                        float* __restrict__ exbuf) {
    int i = blockIdx.x * blockDim.x + threadIdx.x;
    if (i >= (Ee + Nn) * NH) return;
    int e = i >> 2, h = i & 3;
    int d = (e < Ee) ? dst[e] : (e - Ee);
    exbuf[i] = exbuf[i] / (z[d * NH + h] + 1e-16f);
}

// ---------------- weighted scatter (L2-resident f32 atomics) ----------------

__global__ void k_scatter(const int* __restrict__ src, const int* __restrict__ dst,
                          const float* __restrict__ xW, const float* __restrict__ alpha,
                          float* __restrict__ acc) {
    long i = (long)blockIdx.x * blockDim.x + threadIdx.x;  // over (E+N)*256
    if (i >= (long)(Ee + Nn) * KDIM) return;
    int e  = (int)(i >> 8);
    int hc = (int)(i & 255);
    int s, d;
    if (e < Ee) { s = src[e]; d = dst[e]; } else { s = d = e - Ee; }
    float a = alpha[e * NH + (hc >> 6)];
    atomicAdd(&acc[(size_t)d * KDIM + hc], xW[(size_t)s * KDIM + hc] * a);
}

// ---------------- head mean + bias + time embedding ----------------

__global__ void k_mean(const float* __restrict__ acc, const float* __restrict__ b_gat,
                       const float* __restrict__ te_t, float* __restrict__ hs_t) {
    int i = blockIdx.x * blockDim.x + threadIdx.x;      // over N*64
    if (i >= Nn * HIDc) return;
    int c = i & 63;
    const float* p = acc + (size_t)(i >> 6) * KDIM;
    hs_t[i] = 0.25f * (p[c] + p[64 + c] + p[128 + c] + p[192 + c]) + b_gat[c] + te_t[c];
}

// ---------------- fused temporal MHA(last step) + W_out + LayerNorm ----------------

__global__ void __launch_bounds__(128)
k_temporal(const float* __restrict__ hs, const float* __restrict__ Wi,
           const float* __restrict__ bi, const float* __restrict__ Wo,
           const float* __restrict__ bo, const float* __restrict__ W_out,
           const float* __restrict__ b_out, const float* __restrict__ ln_g,
           const float* __restrict__ ln_b, float* __restrict__ out) {
    int n = blockIdx.x, tid = threadIdx.x;              // 128 threads / node

    __shared__ float h[Tt * HIDc];
    __shared__ float kb[Tt * HIDc];
    __shared__ float vb[Tt * HIDc];
    __shared__ float q9[HIDc];
    __shared__ float pr[NH * Tt];
    __shared__ float ob[HIDc];
    __shared__ float attn[HIDc];
    __shared__ float r1[128], r2[128];

    for (int idx = tid; idx < Tt * HIDc; idx += 128) {
        int t = idx >> 6, c = idx & 63;
        h[idx] = hs[(size_t)t * Nn * HIDc + (size_t)n * HIDc + c];
    }
    __syncthreads();

    // q only needed for the last step (output takes attn[:, -1, :])
    if (tid < HIDc) {
        const float* w = Wi + (size_t)tid * HIDc;
        float a = bi[tid];
        const float* hr = h + 9 * HIDc;
#pragma unroll 8
        for (int i = 0; i < HIDc; ++i) a += hr[i] * w[i];
        q9[tid] = a * 0.25f;                            // 1/sqrt(head_dim=16)
    }
    // k, v for all T steps
    for (int idx = tid; idx < Tt * HIDc; idx += 128) {
        int t = idx >> 6, j = idx & 63;
        const float* wk = Wi + (size_t)(HIDc + j) * HIDc;
        const float* wv = Wi + (size_t)(2 * HIDc + j) * HIDc;
        const float* hr = h + t * HIDc;
        float ak = bi[HIDc + j], av = bi[2 * HIDc + j];
#pragma unroll 8
        for (int i = 0; i < HIDc; ++i) { float x = hr[i]; ak += x * wk[i]; av += x * wv[i]; }
        kb[idx] = ak; vb[idx] = av;
    }
    __syncthreads();

    if (tid < NH * Tt) {                                // scores[head][t]
        int hd = tid / Tt, t = tid % Tt;
        float s = 0.f;
#pragma unroll
        for (int d2 = 0; d2 < 16; ++d2) s += q9[hd * 16 + d2] * kb[t * HIDc + hd * 16 + d2];
        pr[hd * Tt + t] = s;
    }
    __syncthreads();

    if (tid < NH) {                                     // softmax over t per head
        float mx = -1e30f;
        for (int t = 0; t < Tt; ++t) mx = fmaxf(mx, pr[tid * Tt + t]);
        float sum = 0.f;
        for (int t = 0; t < Tt; ++t) { float e = __expf(pr[tid * Tt + t] - mx); pr[tid * Tt + t] = e; sum += e; }
        float inv = 1.0f / sum;
        for (int t = 0; t < Tt; ++t) pr[tid * Tt + t] *= inv;
    }
    __syncthreads();

    if (tid < HIDc) {                                   // o = P @ V
        int hd = tid >> 4;
        float s = 0.f;
#pragma unroll
        for (int t = 0; t < Tt; ++t) s += pr[hd * Tt + t] * vb[t * HIDc + tid];
        ob[tid] = s;
    }
    __syncthreads();

    if (tid < HIDc) {                                   // out-projection
        const float* w = Wo + (size_t)tid * HIDc;
        float s = bo[tid];
#pragma unroll 8
        for (int i = 0; i < HIDc; ++i) s += ob[i] * w[i];
        attn[tid] = s;
    }
    __syncthreads();

    // final projection to OUT=128 (one column per thread) + LayerNorm
    float val = b_out[tid];
#pragma unroll 8
    for (int i = 0; i < HIDc; ++i) val += attn[i] * W_out[(size_t)i * OUTc + tid];

    r1[tid] = val; r2[tid] = val * val;
    __syncthreads();
    for (int off = 64; off > 0; off >>= 1) {
        if (tid < off) { r1[tid] += r1[tid + off]; r2[tid] += r2[tid + off]; }
        __syncthreads();
    }
    float mu  = r1[0] * (1.0f / OUTc);
    float var = r2[0] * (1.0f / OUTc) - mu * mu;
    float nv  = (val - mu) * rsqrtf(var + 1e-5f);
    out[(size_t)n * OUTc + tid] = nv * ln_g[tid] + ln_b[tid];
}

// ---------------- host launch ----------------

extern "C" void kernel_launch(void* const* d_in, const int* in_sizes, int n_in,
                              void* d_out, int out_size, void* d_ws, size_t ws_size,
                              hipStream_t stream) {
    (void)in_sizes; (void)n_in; (void)out_size; (void)ws_size;

    const float* x_seq     = (const float*)d_in[0];
    const int*   src       = (const int*)d_in[1];
    const int*   dst       = (const int*)d_in[2];
    const float* W_gat     = (const float*)d_in[3];
    const float* a_src     = (const float*)d_in[4];
    const float* a_dst     = (const float*)d_in[5];
    const float* b_gat     = (const float*)d_in[6];
    const float* time_emb  = (const float*)d_in[7];
    const float* in_proj_w = (const float*)d_in[8];
    const float* in_proj_b = (const float*)d_in[9];
    const float* out_proj_w= (const float*)d_in[10];
    const float* out_proj_b= (const float*)d_in[11];
    const float* W_out     = (const float*)d_in[12];
    const float* b_out     = (const float*)d_in[13];
    const float* ln_g      = (const float*)d_in[14];
    const float* ln_b      = (const float*)d_in[15];

    char* ws = (char*)d_ws;
    size_t off = 0;
    auto alloc = [&](size_t bytes) -> char* {
        char* p = ws + off;
        off += (bytes + 255) & ~(size_t)255;
        return p;
    };

    __bf16* xbf   = (__bf16*)alloc((size_t)Nn * INC * sizeof(__bf16));
    __bf16* WbfT  = (__bf16*)alloc((size_t)KDIM * INC * sizeof(__bf16));
    float*  xW    = (float*)alloc((size_t)Nn * KDIM * sizeof(float));
    float*  al_s  = (float*)alloc((size_t)Nn * NH * sizeof(float));
    float*  al_d  = (float*)alloc((size_t)Nn * NH * sizeof(float));
    float*  z     = (float*)alloc((size_t)Nn * NH * sizeof(float));
    float*  exbuf = (float*)alloc((size_t)(Ee + Nn) * NH * sizeof(float));
    float*  acc   = (float*)alloc((size_t)Nn * KDIM * sizeof(float));
    float*  hsbuf = (float*)alloc((size_t)Tt * Nn * HIDc * sizeof(float));

    k_convW<<<(INC * KDIM + 255) / 256, 256, 0, stream>>>(W_gat, WbfT);

    for (int t = 0; t < Tt; ++t) {
        const int nx = Nn * INC;
        k_convX<<<(nx + 255) / 256, 256, 0, stream>>>(x_seq + (size_t)t * nx, xbf, nx);
        k_init<<<(Nn * KDIM + 255) / 256, 256, 0, stream>>>(acc, z);

        dim3 g(Nn / 16, KDIM / 64);
        k_gemm<<<g, 32, 0, stream>>>(xbf, WbfT, xW);

        k_al<<<(Nn * NH + 255) / 256, 256, 0, stream>>>(xW, a_src, a_dst, al_s, al_d);

        const int ne = (Ee + Nn) * NH;
        const int* src_t = src + (size_t)t * Ee;
        const int* dst_t = dst + (size_t)t * Ee;
        k_esum<<<(ne + 255) / 256, 256, 0, stream>>>(src_t, dst_t, al_s, al_d, exbuf, z);
        k_alpha<<<(ne + 255) / 256, 256, 0, stream>>>(dst_t, z, exbuf);

        long nscat = (long)(Ee + Nn) * KDIM;
        k_scatter<<<(unsigned)((nscat + 255) / 256), 256, 0, stream>>>(src_t, dst_t, xW, exbuf, acc);

        k_mean<<<(Nn * HIDc + 255) / 256, 256, 0, stream>>>(acc, b_gat, time_emb + (size_t)t * HIDc,
                                                            hsbuf + (size_t)t * Nn * HIDc);
    }

    k_temporal<<<Nn, 128, 0, stream>>>(hsbuf, in_proj_w, in_proj_b, out_proj_w, out_proj_b,
                                       W_out, b_out, ln_g, ln_b, (float*)d_out);
}
